// Model_19988777796087
// MI455X (gfx1250) — matrix operand order repaired and checked by reading, not compiled, not run on hardware
//
#include <hip/hip_runtime.h>
#include <hip/hip_bf16.h>

// ---------------------------------------------------------------------------
// Model dims
// ---------------------------------------------------------------------------
#define Bd   8
#define Pd   160
#define Nd   64
#define Dd   384
#define NHd  8
#define HDd  48          // real head dim
#define HDP  64          // padded head dim (multiple of 32 for WMMA K)
#define QKVO (3 * NHd * HDP)   // 1536
#define TOK  (Bd * Pd * Nd)    // 81920
#define PTOK (Bd * Pd)         // 1280
#define NODEc 5672

// gfx1250 async global->LDS staging (GLOBAL_LOAD_ASYNC_TO_LDS_B128 + ASYNCcnt)
#ifndef USE_ASYNC_LDS
#define USE_ASYNC_LDS 1
#endif

typedef __attribute__((ext_vector_type(16))) _Float16 v16h;
typedef __attribute__((ext_vector_type(8)))  float    v8f;
typedef __attribute__((address_space(3)))    _Float16 lds_f16;

enum { ACT_NONE = 0, ACT_GELU = 1, ACT_RELU = 2 };

union FragU { float4 f[2]; v16h h; };

// ---------------------------------------------------------------------------
// WMMA helpers. CDNA5 wave32 layouts (CDNA5 ISA §7.12.2):
//  A 16x32 f16 : lane l (row = l&15), VGPR0..3 = K base..base+7, VGPR4..7 =
//                K base+16..base+23, where base = (l>>4)*8
//  B 32x16 f16 : lane l (col = l&15), same K order as A (base..base+7,
//                base+16..base+23, base = (l>>4)*8)
//  C/D 16x16 f32: lane l col = l&15, VGPR v row = (l>>4)*8 + v
// ---------------------------------------------------------------------------
__device__ __forceinline__ v8f wmma_f16(v16h a, v16h b, v8f c) {
  return __builtin_amdgcn_wmma_f32_16x16x32_f16(false, a, false, b, (short)0, c,
                                                false, false);
}

// A-fragment from row-major f16 buffer (LDS or global), leading dim ld (f16).
__device__ __forceinline__ v16h load_afrag(const _Float16* base, int row0,
                                           int k0, int ld) {
  const int lane = threadIdx.x & 31;
  const _Float16* p =
      base + (size_t)(row0 + (lane & 15)) * ld + k0 + ((lane >> 4) << 3);
  FragU u;
  u.f[0] = *reinterpret_cast<const float4*>(p);        // K base .. base+7
  u.f[1] = *reinterpret_cast<const float4*>(p + 16);   // K base+16 .. base+23
  return u.h;
}

// B-fragment: B[k][n] where source is row-major "n-major": src[n][k] contiguous
// in k (i.e. a transposed weight wT[out][in], or K-rows / V^T rows in LDS).
__device__ __forceinline__ v16h load_bfrag(const _Float16* base, int obase,
                                           int k0, int ld) {
  const int lane = threadIdx.x & 31;
  const _Float16* p =
      base + (size_t)(obase + (lane & 15)) * ld + k0 + ((lane >> 4) << 3);
  FragU u;
  u.f[0] = *reinterpret_cast<const float4*>(p);
  u.f[1] = *reinterpret_cast<const float4*>(p + 16);
  return u.h;
}

__device__ __forceinline__ float gelu_f(float x) {
  float x3 = x * x * x;
  return 0.5f * x * (1.0f + tanhf(0.7978845608028654f * (x + 0.044715f * x3)));
}

// ---------------------------------------------------------------------------
// Generic 64-row GEMM:  C[64 x OUT] = act(A[64 x K] * Bt^T + bias + preadd) + resid
// A: f16 row-major [rows x K]; Bt: f16 [OUT x K] (transposed weights).
// 256 threads = 8 waves; A tile staged into LDS via async global->LDS copies;
// each wave owns chunks of 3 output col-tiles so the 4 A-fragments of a k-step
// stay register-resident across 12 WMMAs.
// ---------------------------------------------------------------------------
template <int K, int OUT, int ACT, bool RESID>
__global__ __launch_bounds__(256) void gemm64(
    const _Float16* __restrict__ A, const _Float16* __restrict__ Bt,
    const float* __restrict__ bias, const float* __restrict__ preadd,
    const float* __restrict__ resid, float* __restrict__ Cf32,
    _Float16* __restrict__ Cf16) {
  constexpr int LDA = K + 8;
  __shared__ _Float16 As[64 * LDA];

  const int row0 = blockIdx.x * 64;
  const int tid  = threadIdx.x;

  // ---- stage A tile (16B chunks) ----
  constexpr int KV8 = K / 8;
#if USE_ASYNC_LDS
  {
    const uint32_t lbase = (uint32_t)(uintptr_t)(lds_f16*)As;
    for (int i = tid; i < 64 * KV8; i += 256) {
      const int r = i / KV8, c = (i % KV8) * 8;
      const _Float16* g = A + (size_t)(row0 + r) * K + c;
      const uint32_t  l = lbase + (uint32_t)(r * LDA + c) * 2u;
      asm volatile("global_load_async_to_lds_b128 %0, %1, off"
                   :: "v"(l), "v"(g) : "memory");
    }
    asm volatile("s_wait_asynccnt 0x0" ::: "memory");
  }
#else
  for (int i = tid; i < 64 * KV8; i += 256) {
    const int r = i / KV8, c = (i % KV8) * 8;
    *reinterpret_cast<float4*>(As + r * LDA + c) =
        *reinterpret_cast<const float4*>(A + (size_t)(row0 + r) * K + c);
  }
#endif
  __syncthreads();

  const int wid  = tid >> 5;
  const int lane = tid & 31;
  const v8f vzero = {0.f, 0.f, 0.f, 0.f, 0.f, 0.f, 0.f, 0.f};

  constexpr int NCT   = OUT / 16;  // 24 or 96 col tiles
  constexpr int CHUNK = 3;         // col tiles per wave per pass

  for (int c0 = wid * CHUNK; c0 < NCT; c0 += 8 * CHUNK) {
    v8f acc[CHUNK][4];
#pragma unroll
    for (int j = 0; j < CHUNK; ++j)
#pragma unroll
      for (int rt = 0; rt < 4; ++rt) acc[j][rt] = vzero;

    // speculative prefetch of the next chunk's weight rows
    if (c0 + 8 * CHUNK < NCT)
      __builtin_prefetch(Bt + (size_t)(c0 + 8 * CHUNK) * 16 * K, 0, 0);

#pragma unroll 2
    for (int ks = 0; ks < K / 32; ++ks) {
      const v16h a0 = load_afrag(As, 0,  ks * 32, LDA);
      const v16h a1 = load_afrag(As, 16, ks * 32, LDA);
      const v16h a2 = load_afrag(As, 32, ks * 32, LDA);
      const v16h a3 = load_afrag(As, 48, ks * 32, LDA);
#pragma unroll
      for (int j = 0; j < CHUNK; ++j) {
        const v16h bf = load_bfrag(Bt, (c0 + j) * 16, ks * 32, K);
        acc[j][0] = wmma_f16(a0, bf, acc[j][0]);
        acc[j][1] = wmma_f16(a1, bf, acc[j][1]);
        acc[j][2] = wmma_f16(a2, bf, acc[j][2]);
        acc[j][3] = wmma_f16(a3, bf, acc[j][3]);
      }
    }

    const int rbase = (lane >> 4) << 3;
#pragma unroll
    for (int j = 0; j < CHUNK; ++j) {
      const int   col = (c0 + j) * 16 + (lane & 15);
      const float bv  = bias ? bias[col] : 0.0f;
#pragma unroll
      for (int rt = 0; rt < 4; ++rt) {
#pragma unroll
        for (int v = 0; v < 8; ++v) {
          const size_t row = (size_t)(row0 + rt * 16 + rbase + v);
          float xv = acc[j][rt][v] + bv;
          if (preadd) xv += preadd[row * OUT + col];
          if constexpr (ACT == ACT_GELU) xv = gelu_f(xv);
          if constexpr (ACT == ACT_RELU) xv = fmaxf(xv, 0.0f);
          if constexpr (RESID) xv += resid[row * OUT + col];
          if (Cf32) Cf32[row * OUT + col] = xv;
          if (Cf16) Cf16[row * OUT + col] = (_Float16)xv;
        }
      }
    }
  }
}

// ---------------------------------------------------------------------------
// Fused attention per (sequence, head, q-block). qkv: f16 [ntok x 1536]
// laid out [3][head][64 padded]; out: f16 [ntok x 384] (heads concatenated,
// 48 real dims per head). MODE 0 = node (mask+bias per patch p = seq%160),
// MODE 1 = patch (bias [P x P], no mask).
// ---------------------------------------------------------------------------
template <int SEQ, int QBLK, int MODE>
__global__ __launch_bounds__(128) void attn_kernel(
    const _Float16* __restrict__ qkv, const int* __restrict__ mask,
    const float* __restrict__ bias, _Float16* __restrict__ out) {
  constexpr int NQ  = SEQ / QBLK;
  constexpr int CTS = SEQ / 16;
  constexpr int RT  = QBLK / 16;

  __shared__ _Float16 q16[QBLK * HDP];
  __shared__ _Float16 k16[SEQ * HDP];
  __shared__ _Float16 vT[HDP * SEQ];
  __shared__ float    s32[QBLK * SEQ];
  __shared__ _Float16 p16[QBLK * SEQ];

  int bidx = blockIdx.x;
  const int qb = bidx % NQ; bidx /= NQ;
  const int h  = bidx % NHd;
  const int s  = bidx / NHd;
  const size_t tokbase = (size_t)s * SEQ;

  const int tid = threadIdx.x, wid = tid >> 5, lane = tid & 31;
  const v8f vzero = {0.f, 0.f, 0.f, 0.f, 0.f, 0.f, 0.f, 0.f};

  for (int i = tid; i < QBLK * HDP; i += 128) {
    int r = i / HDP, c = i % HDP;
    q16[i] = qkv[(tokbase + qb * QBLK + r) * QKVO + h * HDP + c];
  }
  for (int i = tid; i < SEQ * HDP; i += 128) {
    int r = i / HDP, c = i % HDP;
    k16[i]          = qkv[(tokbase + r) * QKVO + 512 + h * HDP + c];
    vT[c * SEQ + r] = qkv[(tokbase + r) * QKVO + 1024 + h * HDP + c];
  }
  __syncthreads();

  // S = Q K^T * scale (+bias, mask)
  const float scale = 0.14433756729740643f;  // 1/sqrt(48)
  for (int idx = wid; idx < RT * CTS; idx += 4) {
    const int rt = idx / CTS, ct = idx % CTS;
    v8f acc = vzero;
#pragma unroll
    for (int ks = 0; ks < HDP / 32; ++ks) {
      v16h a  = load_afrag(q16, rt * 16, ks * 32, HDP);
      v16h bf = load_bfrag(k16, ct * 16, ks * 32, HDP);
      acc = wmma_f16(a, bf, acc);
    }
    const int j  = ct * 16 + (lane & 15);
    const int rb = rt * 16 + ((lane >> 4) << 3);
#pragma unroll
    for (int v = 0; v < 8; ++v) {
      const int iq = qb * QBLK + rb + v;  // query index within sequence
      float val = acc[v] * scale;
      if constexpr (MODE == 0) {
        const int p = s % Pd;
        const size_t bi = ((size_t)p * Nd + iq) * Nd + j;
        val += bias[bi];
        const bool ok = (mask[bi] != 0) || (iq == j);
        if (!ok) val = -1.0e9f;
      } else {
        val += bias[(size_t)iq * Pd + j];
      }
      s32[(rb + v) * SEQ + j] = val;
    }
  }
  __syncthreads();

  // row softmax -> f16 probabilities
  if (tid < QBLK) {
    float m = -3.4e38f;
    for (int j = 0; j < SEQ; ++j) m = fmaxf(m, s32[tid * SEQ + j]);
    float sum = 0.f;
    for (int j = 0; j < SEQ; ++j) {
      float e = __expf(s32[tid * SEQ + j] - m);
      sum += e;
      s32[tid * SEQ + j] = e;
    }
    const float inv = 1.0f / sum;
    for (int j = 0; j < SEQ; ++j)
      p16[tid * SEQ + j] = (_Float16)(s32[tid * SEQ + j] * inv);
  }
  __syncthreads();

  // O = P V  (only the 48 real columns; padded V cols are zero)
  for (int idx = wid; idx < RT * 3; idx += 4) {
    const int rt = idx / 3, ct = idx % 3;
    v8f acc = vzero;
#pragma unroll
    for (int ks = 0; ks < SEQ / 32; ++ks) {
      v16h a  = load_afrag(p16, rt * 16, ks * 32, SEQ);
      v16h bf = load_bfrag(vT, ct * 16, ks * 32, SEQ);
      acc = wmma_f16(a, bf, acc);
    }
    const int col = ct * 16 + (lane & 15);
    const int rb  = rt * 16 + ((lane >> 4) << 3);
#pragma unroll
    for (int v = 0; v < 8; ++v) {
      const size_t row = tokbase + qb * QBLK + rb + v;
      out[row * Dd + h * HDd + col] = (_Float16)acc[v];
    }
  }
}

// ---------------------------------------------------------------------------
// Gather (patch indices, pad row = zeros) + LayerNorm. One wave per token.
// ---------------------------------------------------------------------------
__global__ __launch_bounds__(256) void gather_ln_kernel(
    const float* __restrict__ x, const int* __restrict__ patch,
    float* __restrict__ PX, _Float16* __restrict__ LN16) {
  const int wid = threadIdx.x >> 5, lane = threadIdx.x & 31;
  const int t = blockIdx.x * 8 + wid;
  const int b = t / (Pd * Nd);
  const int r = t % (Pd * Nd);
  const int idx = patch[r];
  const float* src =
      (idx < NODEc) ? (x + ((size_t)b * NODEc + idx) * Dd) : nullptr;
  float vals[12], sum = 0.f, sq = 0.f;
#pragma unroll
  for (int i = 0; i < 12; ++i) {
    const int c = lane + i * 32;
    const float v = src ? src[c] : 0.0f;
    vals[i] = v; sum += v; sq += v * v;
  }
#pragma unroll
  for (int o = 16; o >= 1; o >>= 1) {
    sum += __shfl_xor(sum, o, 32);
    sq  += __shfl_xor(sq, o, 32);
  }
  const float mu  = sum * (1.0f / Dd);
  const float var = sq * (1.0f / Dd) - mu * mu;
  const float inv = rsqrtf(var + 1e-6f);
  float*     pxr = PX + (size_t)t * Dd;
  _Float16*  lnr = LN16 + (size_t)t * Dd;
#pragma unroll
  for (int i = 0; i < 12; ++i) {
    const int c = lane + i * 32;
    pxr[c] = vals[i];
    lnr[c] = (_Float16)((vals[i] - mu) * inv);
  }
}

// LayerNorm rows of f32 [rows x 384] -> f16 (and optional f32 copy of LN out).
__global__ __launch_bounds__(256) void ln_rows_kernel(
    const float* __restrict__ in, _Float16* __restrict__ out16,
    float* __restrict__ out32, int rows) {
  const int wid = threadIdx.x >> 5, lane = threadIdx.x & 31;
  const int r = blockIdx.x * 8 + wid;
  if (r >= rows) return;
  const float* src = in + (size_t)r * Dd;
  float vals[12], sum = 0.f, sq = 0.f;
#pragma unroll
  for (int i = 0; i < 12; ++i) {
    const float v = src[lane + i * 32];
    vals[i] = v; sum += v; sq += v * v;
  }
#pragma unroll
  for (int o = 16; o >= 1; o >>= 1) {
    sum += __shfl_xor(sum, o, 32);
    sq  += __shfl_xor(sq, o, 32);
  }
  const float mu  = sum * (1.0f / Dd);
  const float inv = rsqrtf(sq * (1.0f / Dd) - mu * mu + 1e-6f);
#pragma unroll
  for (int i = 0; i < 12; ++i) {
    const int c = lane + i * 32;
    const float v = (vals[i] - mu) * inv;
    out16[(size_t)r * Dd + c] = (_Float16)v;
    if (out32) out32[(size_t)r * Dd + c] = v;
  }
}

// mean over the N=64 tokens of each (b,p) row
__global__ __launch_bounds__(128) void pmean_kernel(const float* __restrict__ PX,
                                                    float* __restrict__ PM) {
  const int bp = blockIdx.x;
  for (int c = threadIdx.x; c < Dd; c += 128) {
    const float* base = PX + ((size_t)bp * Nd) * Dd + c;
    float s = 0.f;
    for (int n = 0; n < Nd; ++n) s += base[(size_t)n * Dd];
    PM[(size_t)bp * Dd + c] = s * (1.0f / Nd);
  }
}

// wt[o*K + k] = (f16) w[k*OUT + o]
__global__ void transposeW_kernel(const float* __restrict__ w,
                                  _Float16* __restrict__ wt, int K_, int OUT_) {
  const int i = blockIdx.x * 256 + threadIdx.x;
  if (i >= K_ * OUT_) return;
  const int o = i / K_, k = i % K_;
  wt[i] = (_Float16)w[(size_t)k * OUT_ + o];
}

// QKV weights: [D x 1152] -> padded transposed [1536 x 384], layout
// out index = part*512 + head*64 + c (c<48 real, else zero). Bias padded too.
__global__ void qkv_padT_kernel(const float* __restrict__ w,
                                const float* __restrict__ b,
                                _Float16* __restrict__ wt,
                                float* __restrict__ bpad) {
  const int i = blockIdx.x * 256 + threadIdx.x;
  if (i >= 1536 * 384) return;
  const int op = i / 384, k = i % 384;
  const int part = op >> 9, rem = op & 511, head = rem >> 6, c = rem & 63;
  float val = 0.f, bv = 0.f;
  if (c < HDd) {
    const int o = part * Dd + head * HDd + c;
    val = w[(size_t)k * (3 * Dd) + o];
    bv  = b[o];
  }
  wt[i] = (_Float16)val;
  if (k == 0) bpad[op] = bv;
}

__global__ void f32_to_f16_kernel(const float* __restrict__ in,
                                  _Float16* __restrict__ out, int n) {
  const int i = blockIdx.x * 256 + threadIdx.x;
  if (i < n) out[i] = (_Float16)in[i];
}

// broadcast patch vector p2[b*160+p] over the 64 tokens of that patch
__global__ void broadcast_p_kernel(const float* __restrict__ p2,
                                   _Float16* __restrict__ out) {
  const size_t i = (size_t)blockIdx.x * 256 + threadIdx.x;
  if (i >= (size_t)TOK * Dd) return;
  const size_t t = i / Dd;
  const int d = (int)(i % Dd);
  out[i] = (_Float16)p2[(t / Nd) * Dd + d];
}

__global__ void scatter_kernel(const float* __restrict__ PX,
                               const int* __restrict__ patch,
                               float* __restrict__ out) {
  const size_t i = (size_t)blockIdx.x * 256 + threadIdx.x;
  if (i >= (size_t)TOK * Dd) return;
  const size_t t = i / Dd;
  const int d = (int)(i % Dd);
  const int b = (int)(t / (Pd * Nd));
  const int r = (int)(t % (Pd * Nd));
  const int idx = patch[r];
  if (idx < NODEc) out[((size_t)b * NODEc + idx) * Dd + d] = PX[i];
}

// ---------------------------------------------------------------------------
// Host orchestration
// ---------------------------------------------------------------------------
extern "C" void kernel_launch(void* const* d_in, const int* in_sizes, int n_in,
                              void* d_out, int out_size, void* d_ws,
                              size_t ws_size, hipStream_t stream) {
  (void)in_sizes; (void)n_in; (void)out_size; (void)ws_size;

  const float* x        = (const float*)d_in[0];
  const int*   patch    = (const int*)d_in[1];
  const int*   amask    = (const int*)d_in[2];
  const float* abias    = (const float*)d_in[3];
  const float* pab      = (const float*)d_in[4];
  const float* s_qkv_w  = (const float*)d_in[5];
  const float* s_qkv_b  = (const float*)d_in[6];
  const float* s_proj_w = (const float*)d_in[7];
  const float* s_proj_b = (const float*)d_in[8];
  const float* s_fc1_w  = (const float*)d_in[9];
  const float* s_fc1_b  = (const float*)d_in[10];
  const float* s_fc2_w  = (const float*)d_in[11];
  const float* s_fc2_b  = (const float*)d_in[12];
  const float* n_qkv_w  = (const float*)d_in[13];
  const float* n_qkv_b  = (const float*)d_in[14];
  const float* n_proj_w = (const float*)d_in[15];
  const float* n_proj_b = (const float*)d_in[16];
  const float* n_fc1_w  = (const float*)d_in[17];
  const float* n_fc1_b  = (const float*)d_in[18];
  const float* n_fc2_w  = (const float*)d_in[19];
  const float* n_fc2_b  = (const float*)d_in[20];
  const float* fuse_w   = (const float*)d_in[21];
  const float* fuse_b   = (const float*)d_in[22];

  size_t off = 0;
  auto take = [&](size_t bytes) -> void* {
    void* p = (char*)d_ws + off;
    off += (bytes + 255) & ~(size_t)255;
    return p;
  };

  float*     PX     = (float*)take((size_t)TOK * Dd * 4);
  _Float16*  LN16   = (_Float16*)take((size_t)TOK * Dd * 2);
  _Float16*  QKV16  = (_Float16*)take((size_t)TOK * QKVO * 2);
  _Float16*  AO16   = (_Float16*)take((size_t)TOK * Dd * 2);
  _Float16*  sqkvT  = (_Float16*)take((size_t)1536 * 384 * 2);
  float*     sqkvb  = (float*)take(1536 * 4);
  _Float16*  nqkvT  = (_Float16*)take((size_t)1536 * 384 * 2);
  float*     nqkvb  = (float*)take(1536 * 4);
  _Float16*  sprojT = (_Float16*)take((size_t)384 * 384 * 2);
  _Float16*  sfc1T  = (_Float16*)take((size_t)384 * 384 * 2);
  _Float16*  sfc2T  = (_Float16*)take((size_t)384 * 384 * 2);
  _Float16*  nprojT = (_Float16*)take((size_t)384 * 384 * 2);
  _Float16*  nfc1T  = (_Float16*)take((size_t)384 * 384 * 2);
  _Float16*  nfc2T  = (_Float16*)take((size_t)384 * 384 * 2);
  _Float16*  fuseTt = (_Float16*)take((size_t)384 * 384 * 2);
  _Float16*  fuseTb = (_Float16*)take((size_t)384 * 384 * 2);
  float*     PM     = (float*)take((size_t)PTOK * Dd * 4);
  float*     P0     = (float*)take((size_t)PTOK * Dd * 4);
  _Float16*  P16    = (_Float16*)take((size_t)PTOK * Dd * 2);
  _Float16*  QKVP16 = (_Float16*)take((size_t)PTOK * QKVO * 2);
  _Float16*  AOP16  = (_Float16*)take((size_t)PTOK * Dd * 2);
  _Float16*  PL16   = (_Float16*)take((size_t)PTOK * Dd * 2);
  _Float16*  PG16   = (_Float16*)take((size_t)PTOK * Dd * 2);
  // Aliases (safe: originals dead by the time these are used)
  float*     TMP  = (float*)QKV16;   // fuse pass-1 accumulator
  _Float16*  PX16 = AO16;            // f16 copy of PX for fuse pass-1
  _Float16*  A2B  = LN16;            // broadcast patch vectors for fuse pass-2

  const int gW  = (384 * 384 + 255) / 256;
  const int gQW = (1536 * 384 + 255) / 256;
  const int gE  = (int)(((size_t)TOK * Dd + 255) / 256);

  // Weight conversion
  transposeW_kernel<<<gW, 256, 0, stream>>>(s_proj_w, sprojT, 384, 384);
  transposeW_kernel<<<gW, 256, 0, stream>>>(s_fc1_w, sfc1T, 384, 384);
  transposeW_kernel<<<gW, 256, 0, stream>>>(s_fc2_w, sfc2T, 384, 384);
  transposeW_kernel<<<gW, 256, 0, stream>>>(n_proj_w, nprojT, 384, 384);
  transposeW_kernel<<<gW, 256, 0, stream>>>(n_fc1_w, nfc1T, 384, 384);
  transposeW_kernel<<<gW, 256, 0, stream>>>(n_fc2_w, nfc2T, 384, 384);
  transposeW_kernel<<<gW, 256, 0, stream>>>(fuse_w, fuseTt, 384, 384);
  transposeW_kernel<<<gW, 256, 0, stream>>>(fuse_w + 384 * 384, fuseTb, 384, 384);
  qkv_padT_kernel<<<gQW, 256, 0, stream>>>(s_qkv_w, s_qkv_b, sqkvT, sqkvb);
  qkv_padT_kernel<<<gQW, 256, 0, stream>>>(n_qkv_w, n_qkv_b, nqkvT, nqkvb);

  // ---- node level ----
  gather_ln_kernel<<<TOK / 8, 256, 0, stream>>>(x, patch, PX, LN16);
  gemm64<384, 1536, ACT_NONE, false><<<TOK / 64, 256, 0, stream>>>(
      LN16, sqkvT, sqkvb, nullptr, nullptr, nullptr, QKV16);
  attn_kernel<64, 64, 0><<<PTOK * NHd, 128, 0, stream>>>(QKV16, amask, abias, AO16);
  gemm64<384, 384, ACT_NONE, true><<<TOK / 64, 256, 0, stream>>>(
      AO16, sprojT, s_proj_b, nullptr, PX, PX, nullptr);
  ln_rows_kernel<<<TOK / 8, 256, 0, stream>>>(PX, LN16, nullptr, TOK);
  gemm64<384, 384, ACT_GELU, false><<<TOK / 64, 256, 0, stream>>>(
      LN16, sfc1T, s_fc1_b, nullptr, nullptr, nullptr, AO16);
  gemm64<384, 384, ACT_NONE, true><<<TOK / 64, 256, 0, stream>>>(
      AO16, sfc2T, s_fc2_b, nullptr, PX, PX, nullptr);

  // ---- patch level ----
  pmean_kernel<<<PTOK, 128, 0, stream>>>(PX, PM);
  ln_rows_kernel<<<PTOK / 8, 256, 0, stream>>>(PM, P16, P0, PTOK);
  gemm64<384, 1536, ACT_NONE, false><<<PTOK / 64, 256, 0, stream>>>(
      P16, nqkvT, nqkvb, nullptr, nullptr, nullptr, QKVP16);
  attn_kernel<160, 16, 1><<<Bd * NHd * 10, 128, 0, stream>>>(QKVP16, nullptr,
                                                             pab, AOP16);
  gemm64<384, 384, ACT_NONE, true><<<PTOK / 64, 256, 0, stream>>>(
      AOP16, nprojT, n_proj_b, nullptr, P0, P0, nullptr);
  ln_rows_kernel<<<PTOK / 8, 256, 0, stream>>>(P0, PL16, nullptr, PTOK);
  gemm64<384, 384, ACT_GELU, false><<<PTOK / 64, 256, 0, stream>>>(
      PL16, nfc1T, n_fc1_b, nullptr, nullptr, nullptr, PG16);
  gemm64<384, 384, ACT_NONE, true><<<PTOK / 64, 256, 0, stream>>>(
      PG16, nfc2T, n_fc2_b, nullptr, P0, P0, nullptr);

  // ---- fuse: relu([patch_x, p] @ W + b) + patch_x, split over K halves ----
  f32_to_f16_kernel<<<gE, 256, 0, stream>>>(PX, PX16, TOK * Dd);
  gemm64<384, 384, ACT_NONE, false><<<TOK / 64, 256, 0, stream>>>(
      PX16, fuseTt, fuse_b, nullptr, nullptr, TMP, nullptr);
  broadcast_p_kernel<<<gE, 256, 0, stream>>>(P0, A2B);
  gemm64<384, 384, ACT_RELU, true><<<TOK / 64, 256, 0, stream>>>(
      A2B, fuseTb, nullptr, TMP, PX, PX, nullptr);

  // ---- scatter back to node order ----
  scatter_kernel<<<gE, 256, 0, stream>>>(PX, patch, (float*)d_out);
}
